// LstmSimple_53592601919716
// MI455X (gfx1250) — matrix operand order repaired and checked by reading, not compiled, not run on hardware
//
#include <hip/hip_runtime.h>

typedef __bf16 bf16;
typedef __attribute__((ext_vector_type(16))) __bf16 v16bf;
typedef __attribute__((ext_vector_type(8)))  __bf16 v8bf;
typedef __attribute__((ext_vector_type(8)))  float  v8f;

constexpr int Bsz = 512;      // batch
constexpr int Tt  = 1024;     // timesteps
constexpr int In  = 64;       // input dim
constexpr int Hd  = 128;      // hidden dim
constexpr int Kd  = In + Hd;  // 192
constexpr int NC  = 10;       // classes
constexpr int NBLK = 32;      // persistent blocks (8 waves each = 256 waves total)
constexpr int HB  = Bsz * Hd; // elements per h buffer

__device__ __forceinline__ float fast_sigmoid(float z) {
    return __builtin_amdgcn_rcpf(1.0f + __expf(-z));
}
__device__ __forceinline__ float fast_tanh(float z) {
    float e = __expf(-2.0f * z);
    return (1.0f - e) * __builtin_amdgcn_rcpf(1.0f + e);
}

// Device-scope generation barrier over NBLK blocks. Callers pass a strictly
// increasing round number. cnt/gen zeroed by a tiny init kernel each call.
__device__ __forceinline__ void grid_barrier(unsigned* cnt, unsigned* gen, unsigned round) {
    __threadfence();                       // release: push hb stores to L2
    __syncthreads();
    if (threadIdx.x == 0) {
        unsigned prev = __hip_atomic_fetch_add(cnt, 1u, __ATOMIC_ACQ_REL, __HIP_MEMORY_SCOPE_AGENT);
        if (prev == NBLK - 1u) {
            __hip_atomic_store(cnt, 0u, __ATOMIC_RELAXED, __HIP_MEMORY_SCOPE_AGENT);
            __hip_atomic_fetch_add(gen, 1u, __ATOMIC_RELEASE, __HIP_MEMORY_SCOPE_AGENT);
        } else {
            while (__hip_atomic_load(gen, __ATOMIC_ACQUIRE, __HIP_MEMORY_SCOPE_AGENT) < round) {
                __builtin_amdgcn_s_sleep(2);
            }
        }
    }
    __syncthreads();
    __threadfence();                       // acquire: invalidate L0 before reading peers' h
}

__global__ void lstm_barrier_init_kernel(unsigned* bar) {
    bar[0] = 0u;   // cnt
    bar[1] = 0u;   // gen
}

// ---------------------------------------------------------------------------
// Persistent LSTM: 32 blocks x 8 waves = 256 waves; wave owns a fixed
// 16(batch) x 16(hidden) block for all 1024 steps.
//   - Block owns one jTile -> its 4x6 B-fragments (24.6 KB bf16) live in LDS.
//   - B fragments double-buffered in VGPRs: ds_loads for chunk kc+1 overlap
//     the 4 WMMAs of chunk kc (wrapping into the next timestep at kc==5).
//   - Cell state c stays in registers for the whole sequence.
//   - h double-buffered in global ws; one grid barrier per step.
// ---------------------------------------------------------------------------
__global__ __launch_bounds__(256) void lstm_persistent_kernel(
    const float* __restrict__ x,
    const float* __restrict__ Wf, const float* __restrict__ Wo,
    const float* __restrict__ Wi, const float* __restrict__ Wg,
    const float* __restrict__ bfv, const float* __restrict__ bov,
    const float* __restrict__ biv, const float* __restrict__ bgv,
    float* __restrict__ hf, bf16* __restrict__ hb, unsigned* __restrict__ bar)
{
    __shared__ bf16 ldsB[4 * 6 * 32 * 16];            // [q][kc][lane][16] = 24576 B

    const int lane  = threadIdx.x & 31;
    const int wave  = threadIdx.x >> 5;
    const int jTile = blockIdx.x & 7;                 // block owns one jTile
    const int mTile = (blockIdx.x >> 3) * 8 + wave;   // 0..31
    const int m0    = mTile * 16;
    const int j0    = jTile * 16;
    const int lrow  = lane & 15;
    const int g     = lane >> 4;

    // ---- Fill LDS B fragments once (ISA B layout: N = lane%16, K = e<8 ? g*8+e : 16+g*8+e-8) ----
    #pragma unroll
    for (int r = 0; r < 3; ++r) {
        const int combo = wave * 3 + r;               // 24 combos over 8 waves
        const int q = combo / 6, kc = combo % 6;
        const float* W = (q == 0) ? Wf : (q == 1) ? Wo : (q == 2) ? Wi : Wg;
        const float* src = W + (size_t)(j0 + lrow) * Kd + kc * 32 + g * 8;
        bf16* dst = &ldsB[((q * 6 + kc) * 32 + lane) * 16];
        #pragma unroll
        for (int e = 0; e < 16; ++e) dst[e] = (bf16)src[e + (e >= 8 ? 8 : 0)];
    }
    __syncthreads();

    // ---- Zero h buffer 0 (read by t=0), then sync whole grid ----
    #pragma unroll
    for (int v = 0; v < 8; ++v)
        hb[(m0 + v + 8 * g) * Hd + (j0 + lrow)] = (bf16)0.0f;
    unsigned round = 0;
    grid_barrier(&bar[0], &bar[1], ++round);

    // ---- Loop-invariant per-lane values ----
    const int   j  = j0 + lrow;
    const float bF = bfv[j], bO = bov[j], bI = biv[j], bG = bgv[j];
    float creg[8] = {0.f, 0.f, 0.f, 0.f, 0.f, 0.f, 0.f, 0.f};
    const float* xbase = x + (size_t)(m0 + lrow) * Tt * In;

    auto loadB = [&](v16bf* dst, int kc) {
        #pragma unroll
        for (int q = 0; q < 4; ++q)
            dst[q] = *(const v16bf*)&ldsB[((q * 6 + kc) * 32 + lane) * 16];
    };

    // Prime the B pipeline with chunk 0 (fragments are time-invariant).
    v16bf bA[4], bB[4];
    loadB(bA, 0);

    #pragma unroll 1
    for (int t = 0; t < Tt; ++t) {
        const bf16* hprev = hb + (size_t)(t & 1) * HB;
        bf16*       hcur  = hb + (size_t)((t + 1) & 1) * HB;
        const float* xrow = xbase + (size_t)t * In;
        const bf16*  hrow = hprev + (size_t)(m0 + lrow) * Hd;

        // A fragments for all 6 K-chunks up front (hide global latency)
        v16bf af[6];
        #pragma unroll
        for (int kc = 0; kc < 2; ++kc) {              // x region, f32 -> bf16
            const int k0 = kc * 32 + g * 8;
            const int k1 = kc * 32 + 16 + g * 8;
            #pragma unroll
            for (int e = 0; e < 8; ++e) {
                af[kc][e]     = (bf16)xrow[k0 + e];
                af[kc][e + 8] = (bf16)xrow[k1 + e];
            }
        }
        #pragma unroll
        for (int kc = 2; kc < 6; ++kc) {              // h region, bf16 direct
            const int k0 = kc * 32 - 64 + g * 8;
            const int k1 = kc * 32 - 48 + g * 8;
            v8bf s0 = *(const v8bf*)(hrow + k0);
            v8bf s1 = *(const v8bf*)(hrow + k1);
            #pragma unroll
            for (int e = 0; e < 8; ++e) { af[kc][e] = s0[e]; af[kc][e + 8] = s1[e]; }
        }
        __builtin_prefetch(xrow + In, 0, 0);          // next timestep's x row

        // GEMM: 6 chunks x 4 gates; B ping-pongs bA/bB so the ds_loads for
        // chunk kc+1 (mod 6 -> wraps into next t) overlap the WMMAs of kc.
        v8f acc[4] = {};
        #pragma unroll
        for (int kc = 0; kc < 6; ++kc) {
            v16bf* cur = (kc & 1) ? bB : bA;
            v16bf* nxt = (kc & 1) ? bA : bB;
            loadB(nxt, (kc + 1) % 6);
            #pragma unroll
            for (int q = 0; q < 4; ++q) {
                acc[q] = __builtin_amdgcn_wmma_f32_16x16x32_bf16(
                             false, af[kc], false, cur[q], (short)0, acc[q], false, false);
            }
        }

        // Epilogue: activations + in-register cell update
        float hn[8];
        #pragma unroll
        for (int v = 0; v < 8; ++v) {
            float fg = fast_sigmoid(acc[0][v] + bF);
            float og = fast_sigmoid(acc[1][v] + bO);
            float ig = fast_sigmoid(acc[2][v] + bI);
            float gg = fast_tanh(acc[3][v] + bG);
            creg[v] = creg[v] * fg + gg * ig;
            hn[v]   = creg[v] * og;                   // reference: no tanh on c
            hcur[(m0 + v + 8 * g) * Hd + j] = (bf16)hn[v];
        }
        if (t == Tt - 1) {                            // f32 h_T for the FC head
            #pragma unroll
            for (int v = 0; v < 8; ++v)
                hf[(m0 + v + 8 * g) * Hd + j] = hn[v];
        }

        grid_barrier(&bar[0], &bar[1], ++round);
    }
}

// ---------------------------------------------------------------------------
// Final FC: out[b,c] = h_T[b,:] . Wfc[c,:] + bfc[c]   (512x10, trivial)
// ---------------------------------------------------------------------------
__global__ __launch_bounds__(256) void lstm_fc_kernel(
    const float* __restrict__ hf, const float* __restrict__ Wfc,
    const float* __restrict__ bfc, float* __restrict__ out)
{
    int i = blockIdx.x * 256 + threadIdx.x;
    if (i >= Bsz * NC) return;
    int b = i / NC, cc = i % NC;
    const float* hr = hf  + (size_t)b * Hd;
    const float* wr = Wfc + (size_t)cc * Hd;
    float s = bfc[cc];
    #pragma unroll 4
    for (int k = 0; k < Hd; ++k) s += hr[k] * wr[k];
    out[i] = s;
}

extern "C" void kernel_launch(void* const* d_in, const int* in_sizes, int n_in,
                              void* d_out, int out_size, void* d_ws, size_t ws_size,
                              hipStream_t stream) {
    const float* x   = (const float*)d_in[0];
    const float* Wf  = (const float*)d_in[1];
    const float* bf_ = (const float*)d_in[2];
    const float* Wo  = (const float*)d_in[3];
    const float* bo_ = (const float*)d_in[4];
    const float* Wi  = (const float*)d_in[5];
    const float* bi_ = (const float*)d_in[6];
    const float* Wg  = (const float*)d_in[7];
    const float* bg_ = (const float*)d_in[8];
    const float* Wfc = (const float*)d_in[9];
    const float* bfc = (const float*)d_in[10];
    float* out = (float*)d_out;

    // ws layout: hb double buffer (2*131072 B), hf (262144 B), barrier (256 B)
    char*     ws  = (char*)d_ws;
    bf16*     hb  = (bf16*)(ws);
    float*    hf  = (float*)(ws + 262144);
    unsigned* bar = (unsigned*)(ws + 524288);

    hipLaunchKernelGGL(lstm_barrier_init_kernel, dim3(1), dim3(1), 0, stream, bar);
    hipLaunchKernelGGL(lstm_persistent_kernel, dim3(NBLK), dim3(256), 0, stream,
                       x, Wf, Wo, Wi, Wg, bf_, bo_, bi_, bg_, hf, hb, bar);
    hipLaunchKernelGGL(lstm_fc_kernel, dim3(20), dim3(256), 0, stream,
                       hf, Wfc, bfc, out);
}
